// STFT_45818711114408
// MI455X (gfx1250) — matrix operand order repaired and checked by reading, not compiled, not run on hardware
//
#include <hip/hip_runtime.h>
#include <hip/hip_bf16.h>

typedef _Float16 v8h  __attribute__((ext_vector_type(8)));
typedef _Float16 v16h __attribute__((ext_vector_type(16)));
typedef float    v8f  __attribute__((ext_vector_type(8)));

#define NFRAME 512
#define NHOP   128
#define PAD    256
#define BATCH  32
#define TLEN   262144
#define TN     2049          // number of frames
#define CDIM   514           // 2*(NFFT/2+1)
#define OUT_BEG 256          // first padded-coord sample kept (PAD)
#define OUT_END 262400       // one past last kept (out_len - PAD)
#define FT     13            // frames owned per block
#define ROWS   16            // FT + 3 halo frames

// ---------------------------------------------------------------------------
// Kernel 1: fused matrix  M[k,k'] = sum_c weight[c,k] * inv_weight[c,k'],
// written as f16 directly in WMMA B-fragment order:
//   Mshuf[((ks*32 + tile)*32 + lane)*16 + j]  holds  M[K][col]  with
//   K = 32*ks + (lane>=16 ? 16 : 0) + j ,  col = tile*16 + (lane & 15)
// so each lane's 16 halves of a 32x16 B-panel are one contiguous 32B load.
// ---------------------------------------------------------------------------
__global__ __launch_bounds__(256) void stft_fuse_mat(
    const float* __restrict__ w, const float* __restrict__ iw,
    _Float16* __restrict__ mshuf) {
  const int tid  = blockIdx.x * 256 + threadIdx.x;   // 512*128 threads total
  const int K    = tid & 511;                        // coalesced over lanes
  const int col0 = (tid >> 9) * 4;                   // 4 columns per thread
  float s0 = 0.f, s1 = 0.f, s2 = 0.f, s3 = 0.f;
  for (int c = 0; c < CDIM; ++c) {
    const float a = w[c * 512 + K];
    const float* ivr = iw + c * 512 + col0;
    s0 += a * ivr[0]; s1 += a * ivr[1]; s2 += a * ivr[2]; s3 += a * ivr[3];
  }
  const int ks = K >> 5, r = K & 31, hi = r >> 4, j = r & 15;
  const float s[4] = {s0, s1, s2, s3};
  for (int i = 0; i < 4; ++i) {
    const int col  = col0 + i;
    const int lane = hi * 16 + (col & 15);
    const int idx  = ((ks * 32 + (col >> 4)) * 32 + lane) * 16 + j;
    mshuf[idx] = (_Float16)s[i];
  }
}

// ---------------------------------------------------------------------------
// Kernel 2: per block: stage 16 frames (f16) in LDS, Yfr = frames x M via
// chained v_wmma_f32_16x16x32_f16 (two independent accumulator chains per
// wave for XDL pipelining), then deterministic gather overlap-add of the
// owned 13*128 output samples, dividing by the window^2 overlap sum.
// ---------------------------------------------------------------------------
__global__ __launch_bounds__(256) void stft_roundtrip(
    const float* __restrict__ x, const float* __restrict__ window,
    const _Float16* __restrict__ mshuf, float* __restrict__ out) {
  __shared__ _Float16 Af[ROWS][512] __attribute__((aligned(16)));  // 16 KB
  __shared__ float    Yf[ROWS][512];                               // 32 KB
  __shared__ float    w2[NFRAME];                                  //  2 KB

  const int tid = threadIdx.x;
  const int t0  = blockIdx.x * FT;          // first owned frame
  const int b   = blockIdx.y;               // batch

  for (int i = tid; i < NFRAME; i += 256) {
    const float ww = window[i];
    w2[i] = ww * ww;
  }

  // Stage frames t0-3 .. t0+12 (zero outside [0,TN) and outside padded x).
  const float* xb = x + (size_t)b * TLEN;
  for (int e = tid; e < ROWS * 512; e += 256) {
    const int row = e >> 9, k = e & 511;
    const int t  = t0 - 3 + row;
    const int xi = t * NHOP + k - PAD;      // index into unpadded x
    float v = 0.f;
    if (t >= 0 && t < TN && xi >= 0 && xi < TLEN) v = xb[xi];
    Af[row][k] = (_Float16)v;
  }
  __syncthreads();

  const int lane  = tid & 31;
  const int wave  = tid >> 5;
  const int m     = lane & 15;              // A-fragment row
  const int koff  = (lane >> 4) * 16;       // A-fragment K sub-offset
  const int rbase = (lane >> 4) * 8;        // D rows: r (+8 for hi lanes)

  for (int half = 0; half < 2; ++half) {
    const int tile0 = wave + 16 * half;     // two tiles per pass, 8 apart
    const int tile1 = tile0 + 8;
    v8f acc0 = {};
    v8f acc1 = {};
    #pragma unroll
    for (int ks = 0; ks < 16; ++ks) {       // K = 512 in steps of 32
      const int kb = ks * 32;
      // A 16x32 f16 fragment: one contiguous 16-half run (k = 16h + s).
      const v8h a0 = *(const v8h*)&Af[m][kb + koff];
      const v8h a1 = *(const v8h*)&Af[m][kb + koff + 8];
      const v16h a = __builtin_shufflevector(a0, a1, 0, 1, 2, 3, 4, 5, 6, 7,
                                             8, 9, 10, 11, 12, 13, 14, 15);
      // B 32x16 f16 fragments: pre-shuffled, one 32B contiguous load each.
      const v16h b0 = *(const v16h*)(mshuf + ((ks * 32 + tile0) * 32 + lane) * 16);
      const v16h b1 = *(const v16h*)(mshuf + ((ks * 32 + tile1) * 32 + lane) * 16);
      // Two independent WMMA chains -> back-to-back XDL issue, no RAW stall.
      acc0 = __builtin_amdgcn_wmma_f32_16x16x32_f16(
          false, a, false, b0, (short)0, acc0, false, false);
      acc1 = __builtin_amdgcn_wmma_f32_16x16x32_f16(
          false, a, false, b1, (short)0, acc1, false, false);
    }
    const int col0 = tile0 * 16 + (lane & 15);
    const int col1 = tile1 * 16 + (lane & 15);
    #pragma unroll
    for (int r = 0; r < 8; ++r) {
      Yf[rbase + r][col0] = acc0[r];
      Yf[rbase + r][col1] = acc1[r];
    }
  }
  __syncthreads();

  // Deterministic gather overlap-add over the owned output range.
  int nbeg = t0 * NHOP;        if (nbeg < OUT_BEG) nbeg = OUT_BEG;
  int nend = (t0 + FT) * NHOP; if (nend > OUT_END) nend = OUT_END;
  for (int n = nbeg + tid; n < nend; n += 256) {
    const int tl = n >> 7;                  // last frame covering n
    float s = 0.f, cf = 0.f;
    #pragma unroll
    for (int dt = 0; dt < 4; ++dt) {
      const int t = tl - dt;
      if (t >= 0 && t < TN) {
        const int k = (n & 127) + dt * 128;
        s  += Yf[t - t0 + 3][k];
        cf += w2[k];
      }
    }
    out[(size_t)b * TLEN + (n - PAD)] = s / (cf + 1e-8f);
  }
}

// ---------------------------------------------------------------------------
extern "C" void kernel_launch(void* const* d_in, const int* in_sizes, int n_in,
                              void* d_out, int out_size, void* d_ws,
                              size_t ws_size, hipStream_t stream) {
  const float* x          = (const float*)d_in[0];
  const float* weight     = (const float*)d_in[1];
  const float* inv_weight = (const float*)d_in[2];
  const float* window     = (const float*)d_in[3];
  float* out              = (float*)d_out;
  _Float16* mshuf         = (_Float16*)d_ws;     // 512*512 f16 = 512 KB

  // 1) fused 512x512 matrix in WMMA-B order (per launch; deterministic)
  stft_fuse_mat<<<dim3((512 * 128) / 256), dim3(256), 0, stream>>>(
      weight, inv_weight, mshuf);

  // 2) main fused STFT->iSTFT round-trip
  dim3 grid((TN + FT - 1) / FT, BATCH);          // 158 x 32 blocks
  stft_roundtrip<<<grid, dim3(256), 0, stream>>>(x, window, mshuf, out);
}
